// SnConv3_5059471475000
// MI455X (gfx1250) — matrix operand order repaired and checked by reading, not compiled, not run on hardware
//
#include <hip/hip_runtime.h>

typedef __bf16 v16bf __attribute__((ext_vector_type(16)));
typedef float  v8f   __attribute__((ext_vector_type(8)));
typedef unsigned int u32x4 __attribute__((ext_vector_type(4)));

union Frag { v16bf v; u32x4 q[2]; };

__device__ __forceinline__ unsigned short f2bf(float f) {
  unsigned int u = __builtin_bit_cast(unsigned int, f);
  u += 0x7FFFu + ((u >> 16) & 1u);
  return (unsigned short)(u >> 16);
}
__device__ __forceinline__ float bf2f(unsigned short h) {
  unsigned int u = ((unsigned int)h) << 16;
  return __builtin_bit_cast(float, u);
}

// ---------------------------------------------------------------------------
// K1: per (b,i) slab of x[b,i,:,:,:]:
//   - write xbf (bf16 copy of x)
//   - s2[b,i,k,c] = mean_j  -> a2 slot1
//   - s3[b,i,j,c] = mean_k  -> a2 slot2   (LDS f32 atomics for k-reduction)
// a2 element layout: a2bf[((b*32+p)*32+q)*384 + slot*64 + c]
// ---------------------------------------------------------------------------
__global__ __launch_bounds__(256) void k1_s2s3(const float* __restrict__ x,
                                               unsigned short* __restrict__ xbf,
                                               unsigned short* __restrict__ a2bf) {
  __shared__ float s3acc[2048];
  int b = blockIdx.x >> 5, i = blockIdx.x & 31;
  int tid = threadIdx.x;
  for (int e = tid; e < 2048; e += 256) s3acc[e] = 0.f;
  __syncthreads();
  float s2r[8] = {0.f,0.f,0.f,0.f,0.f,0.f,0.f,0.f};
  const float* slab = x + ((size_t)(b*32+i)) * 65536;
  unsigned short* xslab = xbf + ((size_t)(b*32+i)) * 65536;
  int c = tid & 63;
  for (int j = 0; j < 32; ++j) {
    const float* row = slab + j*2048;
    unsigned short* xrow = xslab + j*2048;
    float loc = 0.f;
#pragma unroll
    for (int it = 0; it < 8; ++it) {
      int kc = tid + (it << 8);
      float v = row[kc];
      xrow[kc] = f2bf(v);
      s2r[it] += v;
      loc += v;
    }
    atomicAdd(&s3acc[j*64 + c], loc);
  }
#pragma unroll
  for (int it = 0; it < 8; ++it) {
    int kc = tid + (it << 8);
    int k = kc >> 6;
    a2bf[((size_t)(b*32+i)*32 + k)*384 + 64 + c] = f2bf(s2r[it] * (1.f/32.f));
  }
  __syncthreads();
  for (int e = tid; e < 2048; e += 256) {
    int j = e >> 6, cc = e & 63;
    a2bf[((size_t)(b*32+i)*32 + j)*384 + 128 + cc] = f2bf(s3acc[e] * (1.f/32.f));
  }
}

// ---------------------------------------------------------------------------
// K2: per (b,j): s1[b,j,k,c] = mean_i xbf[b,i,j,k,c] -> a2 slot0
// ---------------------------------------------------------------------------
__global__ __launch_bounds__(256) void k2_s1(const unsigned short* __restrict__ xbf,
                                             unsigned short* __restrict__ a2bf) {
  int b = blockIdx.x >> 5, j = blockIdx.x & 31;
  int tid = threadIdx.x;
  float s1r[8] = {0.f,0.f,0.f,0.f,0.f,0.f,0.f,0.f};
  for (int i = 0; i < 32; ++i) {
    const unsigned short* row = xbf + ((size_t)(b*32+i)*32 + j) * 2048;
#pragma unroll
    for (int it = 0; it < 8; ++it) s1r[it] += bf2f(row[tid + (it << 8)]);
  }
  int c = tid & 63;
#pragma unroll
  for (int it = 0; it < 8; ++it) {
    int kc = tid + (it << 8);
    int k = kc >> 6;
    a2bf[((size_t)(b*32+j)*32 + k)*384 + 0 + c] = f2bf(s1r[it] * (1.f/32.f));
  }
}

// ---------------------------------------------------------------------------
// Kd: diagonals -> a2 slots 3/4/5
//   d12[b,k,d] = x[b,d,d,k]; d13[b,j,d] = x[b,d,j,d]; d23[b,i,d] = x[b,i,d,d]
// thread (b, alpha, c) loops beta.
// ---------------------------------------------------------------------------
__global__ __launch_bounds__(256) void kd_diag(const unsigned short* __restrict__ xbf,
                                               unsigned short* __restrict__ a2bf) {
  int t = blockIdx.x*256 + threadIdx.x;   // 16384
  int b = t >> 11, al = (t >> 6) & 31, c = t & 63;
  for (int be = 0; be < 32; ++be) {
    unsigned short v12 = xbf[((((size_t)b*32+be)*32+be)*32+al)*64 + c];
    unsigned short v13 = xbf[((((size_t)b*32+be)*32+al)*32+be)*64 + c];
    unsigned short v23 = xbf[((((size_t)b*32+al)*32+be)*32+be)*64 + c];
    size_t rb = (((size_t)b*32+al)*32+be)*384;
    a2bf[rb + 192 + c] = v12;
    a2bf[rb + 256 + c] = v13;
    a2bf[rb + 320 + c] = v23;
  }
}

// ---------------------------------------------------------------------------
// K3: all 10 a1 slots derived from a2.  a1f[(b*32+n)*640 + slot*64 + c]
// ---------------------------------------------------------------------------
__global__ __launch_bounds__(256) void k3_a1(const unsigned short* __restrict__ a2bf,
                                             float* __restrict__ a1f) {
  int t = blockIdx.x*256 + threadIdx.x;   // 16384
  int b = t >> 11, n = (t >> 6) & 31, c = t & 63;
  auto meanP = [&](int slot) {
    float s = 0.f;
    for (int p = 0; p < 32; ++p)
      s += bf2f(a2bf[(((size_t)b*32+p)*32+n)*384 + slot*64 + c]);
    return s * (1.f/32.f);
  };
  auto meanQ = [&](int slot) {
    float s = 0.f;
    for (int q = 0; q < 32; ++q)
      s += bf2f(a2bf[(((size_t)b*32+n)*32+q)*384 + slot*64 + c]);
    return s * (1.f/32.f);
  };
  float* o = a1f + (size_t)(b*32+n)*640 + c;
  o[0]   = meanP(0);  // x.mean((1,2))
  o[64]  = meanQ(0);  // x.mean((1,3))
  o[128] = meanQ(1);  // x.mean((2,3))
  o[192] = meanP(3);  // d12.mean(1)
  o[256] = meanQ(3);  // d12.mean(2)
  o[320] = meanP(4);  // d13.mean(1)
  o[384] = meanQ(4);  // d13.mean(2)
  o[448] = meanP(5);  // d23.mean(1)
  o[512] = meanQ(5);  // d23.mean(2)
  o[576] = bf2f(a2bf[(((size_t)b*32+n)*32+n)*384 + 320 + c]); // tdiag
}

// ---------------------------------------------------------------------------
// K3b: a0 (5 slots) from a2.  a0f[b*320 + slot*64 + c]
// ---------------------------------------------------------------------------
__global__ __launch_bounds__(256) void k3b_a0(const unsigned short* __restrict__ a2bf,
                                              float* __restrict__ a0f) {
  int t = blockIdx.x*256 + threadIdx.x;   // 512
  if (t >= 512) return;
  int b = t >> 6, c = t & 63;
  float s0=0.f, s1=0.f, s2=0.f, s3=0.f;
  for (int p = 0; p < 32; ++p)
    for (int q = 0; q < 32; ++q) {
      size_t base = (((size_t)b*32+p)*32+q)*384 + c;
      s0 += bf2f(a2bf[base + 0]);
      s1 += bf2f(a2bf[base + 192]);
      s2 += bf2f(a2bf[base + 256]);
      s3 += bf2f(a2bf[base + 320]);
    }
  float s4 = 0.f;
  for (int d = 0; d < 32; ++d)
    s4 += bf2f(a2bf[(((size_t)b*32+d)*32+d)*384 + 320 + c]);
  float* o = a0f + (size_t)b*320 + c;
  o[0]   = s0 * (1.f/1024.f);
  o[64]  = s1 * (1.f/1024.f);
  o[128] = s2 * (1.f/1024.f);
  o[192] = s3 * (1.f/1024.f);
  o[256] = s4 * (1.f/32.f);
}

// ---------------------------------------------------------------------------
// K4: transpose+convert weights to bf16: wt33[n][k] (64x384), w22t[pl][n][k] (3x64x768)
// ---------------------------------------------------------------------------
__global__ __launch_bounds__(256) void k4_weights(const float* __restrict__ W33,
                                                  const float* __restrict__ W22,
                                                  unsigned short* __restrict__ wt33,
                                                  unsigned short* __restrict__ w22t) {
  int t = blockIdx.x*256 + threadIdx.x;
  if (t < 24576) {
    int n = t / 384, k = t % 384;
    wt33[t] = f2bf(W33[(size_t)k*64 + n]);
  } else if (t < 172032) {
    int u = t - 24576;
    int pl = u / 49152, v = u % 49152;
    int n = v / 768, k = v % 768;
    w22t[u] = f2bf(W22[(size_t)pl*49152 + (size_t)k*64 + n]);
  }
}

// ---------------------------------------------------------------------------
// K5s: small scalar GEMMs: r1, r1t, mdiag, r0 (fp32, bandwidth trivial)
// ---------------------------------------------------------------------------
__global__ __launch_bounds__(256) void k5s_small(const float* __restrict__ a1f, const float* __restrict__ a0f,
                                                 const float* __restrict__ W12, const float* __restrict__ b12,
                                                 const float* __restrict__ W12t, const float* __restrict__ b12t,
                                                 const float* __restrict__ W02, const float* __restrict__ b02,
                                                 const float* __restrict__ W11, const float* __restrict__ b11,
                                                 const float* __restrict__ W01, const float* __restrict__ b01,
                                                 float* __restrict__ r1, float* __restrict__ r1t,
                                                 float* __restrict__ r0, float* __restrict__ mdiag) {
  int t = blockIdx.x*256 + threadIdx.x;
  if (t >= 116224) return;
  if (t < 98304) {                     // r1 / r1t : [3][8][32][64], K=640
    const float* W  = (t < 49152) ? W12 : W12t;
    const float* bi = (t < 49152) ? b12 : b12t;
    float* op       = (t < 49152) ? r1 : r1t;
    int u = t % 49152;
    int pl = u / 16384, rem = u & 16383;
    int bp = rem >> 6, n = rem & 63;
    const float* av = a1f + (size_t)bp*640;
    const float* Wp = W + (size_t)pl*40960;
    float s = bi[pl*64+n];
    for (int z = 0; z < 640; ++z) s += av[z] * Wp[(size_t)z*64 + n];
    op[u] = s;
  } else if (t < 114688) {             // mdiag : [8][32][64]
    int u = t - 98304;
    int bd = u >> 6, n = u & 63;
    const float* av = a1f + (size_t)bd*640;
    const float* a0p = a0f + (size_t)(bd >> 5)*320;
    float s = b11[n] + b01[n];
    for (int z = 0; z < 640; ++z) s += av[z] * W11[(size_t)z*64 + n];
    for (int z = 0; z < 320; ++z) s += a0p[z] * W01[(size_t)z*64 + n];
    mdiag[u] = s;
  } else {                             // r0 : [3][8][64], K=320
    int u = t - 114688;
    int pl = u >> 9, rem = u & 511;
    int b = rem >> 6, n = rem & 63;
    const float* a0p = a0f + (size_t)b*320;
    const float* Wp = W02 + (size_t)pl*20480;
    float s = b02[pl*64+n];
    for (int z = 0; z < 320; ++z) s += a0p[z] * Wp[(size_t)z*64 + n];
    r0[u] = s;
  }
}

// ---------------------------------------------------------------------------
// K5: slices GEMM (WMMA bf16): per plane, [8192 x 768] @ [768 x 64]
//   A row (b,p,q): K<384 -> a2[b,p,q,K] ; K>=384 -> a2[b,q,p,K-384]
//   epilogue adds b22 + r1[p] + r1t[q] + r0, stores slices[pl][b][p][q][n] (f32)
// ---------------------------------------------------------------------------
__global__ __launch_bounds__(256) void k5_slices(const unsigned short* __restrict__ a2bf,
                                                 const unsigned short* __restrict__ w22t,
                                                 const float* __restrict__ b22,
                                                 const float* __restrict__ r1,
                                                 const float* __restrict__ r1t,
                                                 const float* __restrict__ r0,
                                                 float* __restrict__ slices) {
  int wave = threadIdx.x >> 5, lane = threadIdx.x & 31;
  int tg = blockIdx.x*8 + wave;        // 0..1535
  int pl = tg >> 9, t5 = tg & 511;
  int q0 = (t5 & 1) * 16, p = (t5 >> 1) & 31, b = t5 >> 6;
  int lm = lane & 15, hi = lane >> 4, Kb = hi*8;
  int qA = q0 + lm;
  v8f acc[4] = {};
  const u32x4* w4 = (const u32x4*)w22t + (size_t)pl*6144;
  for (int ks = 0; ks < 24; ++ks) {
    int half = (ks >= 12) ? 1 : 0;
    int o1 = ks*32 + Kb - half*384;
    size_t rowb = half ? (((size_t)b*32+qA)*32+p)*384 : (((size_t)b*32+p)*32+qA)*384;
    const u32x4* ap = (const u32x4*)(a2bf + rowb + o1);
    Frag a; a.q[0] = ap[0]; a.q[1] = ap[2];
#pragma unroll
    for (int tc = 0; tc < 4; ++tc) {
      int n = tc*16 + lm;
      const u32x4* bp = w4 + (size_t)n*96 + ks*4 + hi;
      Frag bm; bm.q[0] = bp[0]; bm.q[1] = bp[2];
      acc[tc] = __builtin_amdgcn_wmma_f32_16x16x32_bf16(false, a.v, false, bm.v,
                                                        (short)0, acc[tc], false, false);
    }
  }
  int plb = pl*8 + b;
#pragma unroll
  for (int tc = 0; tc < 4; ++tc) {
    int n = tc*16 + lm;
    float bias = b22[pl*64+n] + r1[((size_t)plb*32+p)*64+n] + r0[(size_t)plb*64+n];
#pragma unroll
    for (int r = 0; r < 8; ++r) {
      int q = q0 + r + hi*8;
      float val = acc[tc][r] + bias + r1t[((size_t)plb*32+q)*64+n];
      slices[(((size_t)plb*32+p)*32+q)*64 + n] = val;
    }
  }
}

// ---------------------------------------------------------------------------
// K6: main GEMM (WMMA bf16): T = allp @ W33 + b33, fused diagonal epilogue.
//   Wave = 16 rows (fixed b,i,j; k in [k0,k0+16)) x 64 out cols.
//   A gathered per permutation straight from bf16 x (L2-resident).
//   W33^T staged in 48KB LDS.
// ---------------------------------------------------------------------------
__global__ __launch_bounds__(256) void k6_main(const unsigned short* __restrict__ xbf,
                                               const unsigned short* __restrict__ wt33,
                                               const float* __restrict__ b33,
                                               const float* __restrict__ slices,
                                               const float* __restrict__ mdiag,
                                               float* __restrict__ out) {
  __shared__ u32x4 wlds[3072];          // 64 x 384 bf16 = 48KB
  int tid = threadIdx.x;
  const u32x4* wg = (const u32x4*)wt33;
  for (int e = tid; e < 3072; e += 256) wlds[e] = wg[e];
  __syncthreads();
  int wave = tid >> 5, lane = tid & 31;
  int t = blockIdx.x*8 + wave;          // 0..16383
  int b = t >> 11, i = (t >> 6) & 31, j = (t >> 1) & 31, k0 = (t & 1)*16;
  int lm = lane & 15, hi = lane >> 4, Kb = hi*8;
  int kl = k0 + lm;
  v8f acc[4] = {};
  for (int ks = 0; ks < 12; ++ks) {
    int pm = ks >> 1;
    int cb = (ks & 1)*32 + Kb;
    int A1, A2, A3;
    switch (pm) {       // itertools.permutations((1,2,3)) order
      case 0: A1=i;  A2=j;  A3=kl; break;   // (1,2,3)
      case 1: A1=i;  A2=kl; A3=j;  break;   // (1,3,2)
      case 2: A1=j;  A2=i;  A3=kl; break;   // (2,1,3)
      case 3: A1=kl; A2=i;  A3=j;  break;   // (2,3,1)
      case 4: A1=j;  A2=kl; A3=i;  break;   // (3,1,2)
      default:A1=kl; A2=j;  A3=i;  break;   // (3,2,1)
    }
    size_t off = ((((size_t)b*32+A1)*32+A2)*32+A3)*64 + cb;
    const u32x4* ap = (const u32x4*)(xbf + off);
    Frag a; a.q[0] = ap[0]; a.q[1] = ap[2];
#pragma unroll
    for (int tc = 0; tc < 4; ++tc) {
      int n = tc*16 + lm;
      const u32x4* bp = &wlds[n*48 + ks*4 + hi];
      Frag bm; bm.q[0] = bp[0]; bm.q[1] = bp[2];
      acc[tc] = __builtin_amdgcn_wmma_f32_16x16x32_bf16(false, a.v, false, bm.v,
                                                        (short)0, acc[tc], false, false);
    }
  }
#pragma unroll
  for (int tc = 0; tc < 4; ++tc) {
    int n = tc*16 + lm;
    float bias = b33[n];
#pragma unroll
    for (int r = 0; r < 8; ++r) {
      int kk = k0 + r + hi*8;
      float val = acc[tc][r] + bias;
      if (i == j)  val += slices[((size_t)(0*8+b)*1024 + (size_t)kk*32 + i)*64 + n];
      if (i == kk) val += slices[((size_t)(1*8+b)*1024 + (size_t)j*32  + i)*64 + n];
      if (j == kk) val += slices[((size_t)(2*8+b)*1024 + (size_t)i*32  + j)*64 + n];
      if (i == j && i == kk) val += mdiag[((size_t)b*32 + i)*64 + n];
      out[((((size_t)b*32+i)*32+j)*32+kk)*64 + n] = val;
    }
  }
}

// ---------------------------------------------------------------------------
extern "C" void kernel_launch(void* const* d_in, const int* in_sizes, int n_in,
                              void* d_out, int out_size, void* d_ws, size_t ws_size,
                              hipStream_t stream) {
  const float* x    = (const float*)d_in[0];
  const float* W33  = (const float*)d_in[1];
  const float* b33  = (const float*)d_in[2];
  const float* W22  = (const float*)d_in[3];
  const float* b22  = (const float*)d_in[4];
  const float* W12  = (const float*)d_in[5];
  const float* b12  = (const float*)d_in[6];
  const float* W12t = (const float*)d_in[7];
  const float* b12t = (const float*)d_in[8];
  const float* W02  = (const float*)d_in[9];
  const float* b02  = (const float*)d_in[10];
  const float* W11  = (const float*)d_in[11];
  const float* b11  = (const float*)d_in[12];
  const float* W01  = (const float*)d_in[13];
  const float* b01  = (const float*)d_in[14];
  float* out = (float*)d_out;

  char* ws = (char*)d_ws;
  unsigned short* xbf  = (unsigned short*)(ws);              // 33,554,432 B
  unsigned short* a2bf = (unsigned short*)(ws + 33554432);   //  6,291,456 B
  float* a1f           = (float*)(ws + 39845888);            //    655,360 B
  float* a0f           = (float*)(ws + 40501248);            //     10,240 B
  unsigned short* wt33 = (unsigned short*)(ws + 40511488);   //     49,152 B
  unsigned short* w22t = (unsigned short*)(ws + 40560640);   //    294,912 B
  float* r1            = (float*)(ws + 40855552);            //    196,608 B
  float* r1t           = (float*)(ws + 41052160);            //    196,608 B
  float* r0            = (float*)(ws + 41248768);            //      6,144 B
  float* md            = (float*)(ws + 41254912);            //     65,536 B
  float* slices        = (float*)(ws + 41320448);            //  6,291,456 B  (~45.4 MB total)

  k1_s2s3  <<<256,  256, 0, stream>>>(x, xbf, a2bf);
  k2_s1    <<<256,  256, 0, stream>>>(xbf, a2bf);
  kd_diag  <<<64,   256, 0, stream>>>(xbf, a2bf);
  k3_a1    <<<64,   256, 0, stream>>>(a2bf, a1f);
  k3b_a0   <<<2,    256, 0, stream>>>(a2bf, a0f);
  k4_weights<<<672, 256, 0, stream>>>(W33, W22, wt33, w22t);
  k5s_small<<<454,  256, 0, stream>>>(a1f, a0f, W12, b12, W12t, b12t,
                                      W02, b02, W11, b11, W01, b01,
                                      r1, r1t, r0, md);
  k5_slices<<<192,  256, 0, stream>>>(a2bf, w22t, b22, r1, r1t, r0, slices);
  k6_main  <<<2048, 256, 0, stream>>>(xbf, wt33, b33, slices, md, out);
  (void)in_sizes; (void)n_in; (void)out_size; (void)ws_size;
}
